// CA_84997402788103
// MI455X (gfx1250) — matrix operand order repaired and checked by reading, not compiled, not run on hardware
//
#include <hip/hip_runtime.h>

#define CHN 12
#define HID 96
#define IMG 512
#define HW  (IMG * IMG)

typedef __attribute__((ext_vector_type(16))) _Float16 v16h;
typedef __attribute__((ext_vector_type(8)))  float    v8f;

// One workgroup (128 threads = 4 waves) processes one (batch, row) pair.
// Each wave handles 8 tiles of 16 consecutive pixels along the row.
__launch_bounds__(128, 1)
__global__ void nca_step_kernel(const float* __restrict__ x,
                                const float* __restrict__ rand_u,
                                const float* __restrict__ w1_w,
                                const float* __restrict__ w1_b,
                                const float* __restrict__ w2_w,
                                float* __restrict__ out)
{
    __shared__ float sW1[HID * 48];
    __shared__ float sB1[HID];
    __shared__ float sW2[CHN * HID];

    const int tid = threadIdx.x;
    for (int i = tid; i < HID * 48; i += 128) sW1[i] = w1_w[i];
    for (int i = tid; i < HID;      i += 128) sB1[i] = w1_b[i];
    for (int i = tid; i < CHN * HID; i += 128) sW2[i] = w2_w[i];
    __syncthreads();

    const int lane = tid & 31;
    const int wave = tid >> 5;
    const int lhi  = lane >> 4;     // 0: lanes 0-15, 1: lanes 16-31
    const int l16  = lane & 15;

    const int b   = blockIdx.x >> 9;      // image index (8)
    const int y   = blockIdx.x & (IMG-1); // row (512)
    const int ym1 = (y + IMG - 1) & (IMG - 1);
    const int yp1 = (y + 1) & (IMG - 1);

    // ---------------- Resident A operands (weights), WMMA A layout ----------------
    // 16x32 f16 A tile: lane holds row M = 16*mt + (lane&15);
    // half h -> K = 32*ks + (lhi?16:0) + h
    v16h A1[6][2];
    #pragma unroll
    for (int mt = 0; mt < 6; ++mt) {
        const int M = 16 * mt + l16;
        #pragma unroll
        for (int ks = 0; ks < 2; ++ks) {
            v16h a;
            #pragma unroll
            for (int h = 0; h < 16; ++h) {
                const int K = 32 * ks + (lhi ? 16 : 0) + h;
                float v = 0.0f;
                if (K < 48)       v = sW1[M * 48 + K];
                else if (K == 48) v = sB1[M];     // bias folded into padded K column
                a[h] = (_Float16)v;
            }
            A1[mt][ks] = a;
        }
    }
    v16h A2[3];   // W2 padded to 16 output rows, K = 0..95
    #pragma unroll
    for (int ks = 0; ks < 3; ++ks) {
        v16h a;
        #pragma unroll
        for (int h = 0; h < 16; ++h) {
            const int K = 32 * ks + (lhi ? 16 : 0) + h;
            const float v = (l16 < CHN) ? sW2[l16 * HID + K] : 0.0f;
            a[h] = (_Float16)v;
        }
        A2[ks] = a;
    }

    // ---------------- Pixel tile loop ----------------
    for (int t = 0; t < 8; ++t) {
        const int px  = wave * 128 + t * 16 + l16;
        const int xm1 = (px + IMG - 1) & (IMG - 1);
        const int xp1 = (px + 1) & (IMG - 1);

        // B1 (32x16 f16, K = perception channel): lane = column N = pixel,
        // half h -> K = 32*ks + (lhi?16:0) + h. Perception channel 4c+f.
        v16h B1[2];
        #pragma unroll
        for (int ks = 0; ks < 2; ++ks) {
            v16h bb;
            const int c0 = 8 * ks + (lhi ? 4 : 0);   // first of 4 input channels
            #pragma unroll
            for (int j = 0; j < 4; ++j) {
                const int c = c0 + j;
                float ident = 0.f, sx = 0.f, sy = 0.f, lap = 0.f;
                if (c < CHN) {
                    const float* xc = x + ((size_t)(b * CHN + c)) * HW;
                    const float a00 = xc[ym1 * IMG + xm1], a01 = xc[ym1 * IMG + px], a02 = xc[ym1 * IMG + xp1];
                    const float a10 = xc[y   * IMG + xm1], a11 = xc[y   * IMG + px], a12 = xc[y   * IMG + xp1];
                    const float a20 = xc[yp1 * IMG + xm1], a21 = xc[yp1 * IMG + px], a22 = xc[yp1 * IMG + xp1];
                    ident = a11;
                    sx  = (a02 - a00) + 2.0f * (a12 - a10) + (a22 - a20);
                    sy  = (a20 - a00) + 2.0f * (a21 - a01) + (a22 - a02);
                    lap = a00 + 2.0f*a01 + a02 + 2.0f*a10 - 12.0f*a11 + 2.0f*a12 + a20 + 2.0f*a21 + a22;
                }
                bb[4*j + 0] = (_Float16)ident;
                bb[4*j + 1] = (_Float16)sx;
                bb[4*j + 2] = (_Float16)sy;
                bb[4*j + 3] = (_Float16)lap;
            }
            B1[ks] = bb;
        }
        // bias column K==48 lives at ks=1, hi-lane half 0
        {
            v16h bb = B1[1];
            bb[0] = lhi ? (_Float16)1.0f : bb[0];
            B1[1] = bb;
        }

        // -------- GEMM1: H = W1 * Y (+bias), then ReLU --------
        v8f acc[6];
        #pragma unroll
        for (int mt = 0; mt < 6; ++mt) {
            v8f c = {0.f,0.f,0.f,0.f,0.f,0.f,0.f,0.f};
            c = __builtin_amdgcn_wmma_f32_16x16x32_f16(false, A1[mt][0], false, B1[0], (short)0, c, false, false);
            c = __builtin_amdgcn_wmma_f32_16x16x32_f16(false, A1[mt][1], false, B1[1], (short)0, c, false, false);
            #pragma unroll
            for (int r = 0; r < 8; ++r) c[r] = fmaxf(c[r], 0.0f);
            acc[mt] = c;
        }

        // -------- GEMM2: O = W2 * H --------
        // Convert accumulator (C/D layout: lane=N, vgpr r -> M = r + 8*lhi) into
        // B layout (lane=N, half h -> K = 32*ks + 16*lhi + h). N = lane&15 matches
        // in both; only the M&8 half must swap between lane l and l^16.
        v8f oacc = {0.f,0.f,0.f,0.f,0.f,0.f,0.f,0.f};
        #pragma unroll
        for (int ks = 0; ks < 3; ++ks) {
            const v8f ha = acc[2*ks + 0];
            const v8f hb = acc[2*ks + 1];
            v16h bb;
            #pragma unroll
            for (int r = 0; r < 8; ++r) {
                const float send = lhi ? ha[r] : hb[r];
                const float recv = __shfl_xor(send, 16, 32);
                const float own  = lhi ? hb[r] : ha[r];
                bb[r]     = (_Float16)(lhi ? recv : own);   // K = 32ks + 16*lhi + r
                bb[8 + r] = (_Float16)(lhi ? own  : recv);  // K = 32ks + 16*lhi + 8 + r
            }
            oacc = __builtin_amdgcn_wmma_f32_16x16x32_f16(false, A2[ks], false, bb, (short)0, oacc, false, false);
        }

        // -------- mask, residual, store --------
        const float ru   = rand_u[(size_t)b * HW + (size_t)y * IMG + px];
        const float mask = floorf(ru + 0.5f);
        #pragma unroll
        for (int r = 0; r < 8; ++r) {
            const int M = r + lhi * 8;       // output channel of this accumulator row
            if (M < CHN) {
                const size_t idx = ((size_t)(b * CHN + M)) * HW + (size_t)y * IMG + px;
                out[idx] = x[idx] + oacc[r] * mask;
            }
        }
    }
}

extern "C" void kernel_launch(void* const* d_in, const int* in_sizes, int n_in,
                              void* d_out, int out_size, void* d_ws, size_t ws_size,
                              hipStream_t stream) {
    (void)in_sizes; (void)n_in; (void)out_size; (void)d_ws; (void)ws_size;
    const float* x      = (const float*)d_in[0];  // [8,12,512,512]
    const float* rand_u = (const float*)d_in[1];  // [8,1,512,512]
    const float* w1_w   = (const float*)d_in[2];  // [96,48]
    const float* w1_b   = (const float*)d_in[3];  // [96]
    const float* w2_w   = (const float*)d_in[4];  // [12,96]
    float* out = (float*)d_out;                   // [8,12,512,512]

    dim3 grid(8 * IMG);   // one block per (batch, row)
    dim3 block(128);      // 4 waves
    nca_step_kernel<<<grid, block, 0, stream>>>(x, rand_u, w1_w, w1_b, w2_w, out);
}
